// MFABlock_22857815949515
// MI455X (gfx1250) — matrix operand order repaired and checked by reading, not compiled, not run on hardware
//
#include <hip/hip_runtime.h>
#include <hip/hip_bf16.h>

// ---------------- problem constants ----------------
#define DIMC 96      // model dim
#define DINC 192     // d_inner
#define DSN  16      // d_state
#define DTRK 6       // dt_rank
#define KCONV 4      // causal conv taps
#define NB   2       // batch
#define EDGE 24      // H = W = Z
#define LT   13824   // EDGE^3
#define BLT  27648   // NB * LT
#define XLD  48      // padded DTR+2*DS (=38) to 3 N-tiles

typedef __attribute__((ext_vector_type(16))) _Float16 v16h;
typedef __attribute__((ext_vector_type(8)))  float    v8f;

// ---------------- WMMA helpers (gfx1250 wave32, 16x16x32 f16) ----------------
// A (16x32 f16) lane layout: lane = hi*16 + r, half j ->
//   K = hi*16 + j, row M = r
__device__ __forceinline__ int kmap(int j, int hi) {
  return hi * 16 + j;
}

__device__ __forceinline__ v8f wmma16(v16h a, v16h b, v8f c) {
  return __builtin_amdgcn_wmma_f32_16x16x32_f16(false, a, false, b, (short)0, c,
                                                false, false);
}

// A fragment from row-major f16 (row stride lda)
__device__ __forceinline__ v16h frag_a_rm(const _Float16* A, long lda, long row0,
                                          int k0, int lane) {
  int r = lane & 15, hi = lane >> 4;
  v16h f;
#pragma unroll
  for (int j = 0; j < 16; ++j) f[j] = A[(row0 + r) * lda + k0 + kmap(j, hi)];
  return f;
}

// B fragment from weight-style (N x K) row-major: B[k][n] = W[n*ldw + k]
// B lane layout: lane = hi*16 + n, half j -> K = hi*16 + j, col N = n
__device__ __forceinline__ v16h frag_b_wt(const _Float16* W, long ldw, int n0,
                                          int k0, int lane) {
  int n = lane & 15, hi = lane >> 4;
  v16h f;
#pragma unroll
  for (int j = 0; j < 16; ++j) f[j] = W[(long)(n0 + n) * ldw + k0 + hi * 16 + j];
  return f;
}

// B fragment from k-major (K x N) row-major
__device__ __forceinline__ v16h frag_b_km(const _Float16* M, long ldb, int k0,
                                          int n0, int lane) {
  int n = lane & 15, hi = lane >> 4;
  v16h f;
#pragma unroll
  for (int j = 0; j < 16; ++j) f[j] = M[(long)(k0 + hi * 16 + j) * ldb + n0 + n];
  return f;
}

__device__ __forceinline__ float siluf(float x) { return x / (1.f + __expf(-x)); }
__device__ __forceinline__ float softplusf(float x) {
  return (x > 20.f) ? x : log1pf(__expf(x));
}

// scan-position l -> original token index, per branch permutation
__device__ __forceinline__ int perm_idx(int br, int l) {
  if (br == 0) return l;
  if (br == 1) return LT - 1 - l;
  const int q = LT / 4;
  int s = l / q;
  int j = l - s * q;
  return j * 4 + s;
}

// ---------------- packing kernels ----------------
__global__ void k_cast_f16(const float* __restrict__ s, _Float16* __restrict__ d,
                           int n) {
  int i = blockIdx.x * blockDim.x + threadIdx.x;
  if (i < n) d[i] = (_Float16)s[i];
}

__global__ void k_pack_xproj(const float* __restrict__ w, _Float16* __restrict__ d) {
  int i = blockIdx.x * blockDim.x + threadIdx.x;
  if (i >= 3 * XLD * DINC) return;
  int br = i / (XLD * DINC);
  int rem = i - br * XLD * DINC;
  int n = rem / DINC, k = rem % DINC;
  d[i] = (n < DTRK + 2 * DSN)
             ? (_Float16)w[((size_t)br * (DTRK + 2 * DSN) + n) * DINC + k]
             : (_Float16)0.f;
}

// f_w (OC, IC, 3,3,3) -> (OC, 27, IC) f16
__global__ void k_pack_convw(const float* __restrict__ w, _Float16* __restrict__ d) {
  int i = blockIdx.x * blockDim.x + threadIdx.x;
  if (i >= DIMC * 27 * DINC) return;
  int oc = i / (27 * DINC);
  int rem = i - oc * 27 * DINC;
  int tap = rem / DINC, ic = rem % DINC;
  d[i] = (_Float16)w[((size_t)oc * DINC + ic) * 27 + tap];
}

__global__ void k_pack_aneg(const float* __restrict__ alog, float* __restrict__ d) {
  int i = blockIdx.x * blockDim.x + threadIdx.x;
  if (i < 3 * DINC * DSN) d[i] = -__expf(alog[i]);
}

// ---------------- layernorm over channels (x is (B, C, L)) ----------------
__global__ void k_layernorm(const float* __restrict__ x, const float* __restrict__ g,
                            const float* __restrict__ b, _Float16* __restrict__ o) {
  int t = blockIdx.x * blockDim.x + threadIdx.x;
  if (t >= BLT) return;
  int bb = t / LT, l = t - bb * LT;
  const float* xb = x + (size_t)bb * DIMC * LT + l;
  float mu = 0.f;
  for (int c = 0; c < DIMC; ++c) mu += xb[(size_t)c * LT];
  mu *= (1.f / DIMC);
  float var = 0.f;
  for (int c = 0; c < DIMC; ++c) {
    float dd = xb[(size_t)c * LT] - mu;
    var += dd * dd;
  }
  var *= (1.f / DIMC);
  float inv = rsqrtf(var + 1e-5f);
  for (int c = 0; c < DIMC; ++c)
    o[(size_t)t * DIMC + c] = (_Float16)((xb[(size_t)c * LT] - mu) * inv * g[c] + b[c]);
}

// ---------------- in_proj GEMM: (BLT x 96) @ (96 x 384) ----------------
__global__ void k_gemm_inproj(const _Float16* __restrict__ a16,
                              const _Float16* __restrict__ w16,
                              float* __restrict__ xin, float* __restrict__ gate) {
  int wave = (blockIdx.x * blockDim.x + threadIdx.x) >> 5;
  int lane = threadIdx.x & 31;
  if (wave >= (BLT / 16) * 24) return;
  int tm = wave / 24, tn = wave % 24;
  v8f acc = {};
#pragma unroll
  for (int k0 = 0; k0 < DIMC; k0 += 32) {
    v16h a = frag_a_rm(a16, DIMC, (long)tm * 16, k0, lane);
    v16h bf = frag_b_wt(w16, DIMC, tn * 16, k0, lane);
    acc = wmma16(a, bf, acc);
  }
  int n = lane & 15, hi = lane >> 4;
#pragma unroll
  for (int v = 0; v < 8; ++v) {
    size_t row = (size_t)tm * 16 + v + hi * 8;
    int col = tn * 16 + n;
    float val = acc[v];
    if (col < DINC) xin[row * DINC + col] = val;
    else            gate[row * DINC + (col - DINC)] = siluf(val);
  }
}

// ---------------- depthwise causal conv1d (+silu), permuted gather ----------
__global__ void k_conv1d(const float* __restrict__ xin, const float* __restrict__ cw,
                         const float* __restrict__ cb, float* __restrict__ u,
                         _Float16* __restrict__ u16, int br) {
  int i = blockIdx.x * blockDim.x + threadIdx.x;
  if (i >= BLT * DINC) return;
  int d = i % DINC;
  int t = i / DINC;          // b*LT + l  (scan domain)
  int b = t / LT, l = t - b * LT;
  float acc = cb[br * DINC + d];
  const float* w = cw + (size_t)(br * DINC + d) * KCONV;
#pragma unroll
  for (int k = 0; k < KCONV; ++k) {
    int ls = l - (KCONV - 1) + k;
    if (ls >= 0)
      acc += w[k] * xin[((size_t)b * LT + perm_idx(br, ls)) * DINC + d];
  }
  float uu = siluf(acc);
  u[i] = uu;
  u16[i] = (_Float16)uu;
}

// ---------------- xproj GEMM: (BLT x 192) @ (192 x 48pad) ----------------
__global__ void k_gemm_xproj(const _Float16* __restrict__ u16,
                             const _Float16* __restrict__ w16,
                             float* __restrict__ xdbl) {
  int wave = (blockIdx.x * blockDim.x + threadIdx.x) >> 5;
  int lane = threadIdx.x & 31;
  if (wave >= (BLT / 16) * 3) return;
  int tm = wave / 3, tn = wave % 3;
  v8f acc = {};
#pragma unroll
  for (int k0 = 0; k0 < DINC; k0 += 32) {
    v16h a = frag_a_rm(u16, DINC, (long)tm * 16, k0, lane);
    v16h bf = frag_b_wt(w16, DINC, tn * 16, k0, lane);
    acc = wmma16(a, bf, acc);
  }
  int n = lane & 15, hi = lane >> 4;
#pragma unroll
  for (int v = 0; v < 8; ++v)
    xdbl[((size_t)tm * 16 + v + hi * 8) * XLD + tn * 16 + n] = acc[v];
}

// ---------------- dt = softplus(dtr @ dt_w^T + dt_b), K=6 (VALU) ----------
__global__ void k_dtproj(const float* __restrict__ xdbl, const float* __restrict__ dtw,
                         const float* __restrict__ dtb, float* __restrict__ dt, int br) {
  int i = blockIdx.x * blockDim.x + threadIdx.x;
  if (i >= BLT * DINC) return;
  int d = i % DINC;
  int t = i / DINC;
  float a = dtb[br * DINC + d];
  const float* w = dtw + (size_t)(br * DINC + d) * DTRK;
  const float* xr = xdbl + (size_t)t * XLD;
#pragma unroll
  for (int r = 0; r < DTRK; ++r) a += w[r] * xr[r];
  dt[i] = softplusf(a);
}

// ---------------- selective scan: lane per (b,d,s), shuffle-reduce over s ---
__global__ void k_scan(const float* __restrict__ u, const float* __restrict__ dt,
                       const float* __restrict__ xdbl, const float* __restrict__ aneg,
                       const float* __restrict__ Dp, const float* __restrict__ gate,
                       _Float16* __restrict__ y16, int br) {
  int wave = (blockIdx.x * blockDim.x + threadIdx.x) >> 5;
  int lane = threadIdx.x & 31;
  int s = lane & 15;
  int pair = wave * 2 + (lane >> 4);   // (b,d) channel
  if (pair >= NB * DINC) return;
  int b = pair / DINC, d = pair - b * DINC;
  float Av = aneg[(size_t)(br * DINC + d) * DSN + s];
  float Dv = Dp[br * DINC + d];
  float h = 0.f;
  for (int l = 0; l < LT; ++l) {
    size_t t = (size_t)b * LT + l;
    if (l + 16 < LT) __builtin_prefetch(&dt[(t + 16) * DINC + d], 0, 0);
    float dtv = dt[t * DINC + d];
    float uu  = u[t * DINC + d];
    float Bc  = xdbl[t * XLD + DTRK + s];
    float Cc  = xdbl[t * XLD + DTRK + DSN + s];
    float dA = __expf(dtv * Av);
    h = dA * h + dtv * uu * Bc;
    float y = h * Cc;
    y += __shfl_xor(y, 1, 16);
    y += __shfl_xor(y, 2, 16);
    y += __shfl_xor(y, 4, 16);
    y += __shfl_xor(y, 8, 16);
    if (s == 0) {
      int o = perm_idx(br, l);                       // scatter through inverse perm
      size_t oi = ((size_t)b * LT + o) * DINC + d;
      y16[oi] = (_Float16)((y + uu * Dv) * gate[oi]);
    }
  }
}

__global__ void k_sum_qkv(const _Float16* __restrict__ q, const _Float16* __restrict__ k,
                          const _Float16* __restrict__ v, _Float16* __restrict__ o) {
  int i = blockIdx.x * blockDim.x + threadIdx.x;
  if (i < BLT * DINC)
    o[i] = (_Float16)((float)q[i] + (float)k[i] + (float)v[i]);
}

// ---------------- out_proj GEMM -> channel-major f16 (conv2 input hi half) --
__global__ void k_gemm_out(const _Float16* __restrict__ a16,
                           const _Float16* __restrict__ w16,
                           _Float16* __restrict__ conv2in) {
  int wave = (blockIdx.x * blockDim.x + threadIdx.x) >> 5;
  int lane = threadIdx.x & 31;
  if (wave >= (BLT / 16) * 6) return;
  int tm = wave / 6, tn = wave % 6;
  v8f acc = {};
#pragma unroll
  for (int k0 = 0; k0 < DINC; k0 += 32) {
    v16h a = frag_a_rm(a16, DINC, (long)tm * 16, k0, lane);
    v16h bf = frag_b_wt(w16, DINC, tn * 16, k0, lane);
    acc = wmma16(a, bf, acc);
  }
  int n = lane & 15, hi = lane >> 4;
#pragma unroll
  for (int v = 0; v < 8; ++v) {
    int row = tm * 16 + v + hi * 8;
    int b = row / LT, l = row - b * LT;
    int oc = tn * 16 + n;
    conv2in[(size_t)b * DINC * LT + (size_t)(DIMC + oc) * LT + l] = (_Float16)acc[v];
  }
}

// ---------------- attention logits: S[b,d,e] = sum_l q[l,d]*k[l,e] ----------
__global__ void k_gemm_attn(const _Float16* __restrict__ q16,
                            const _Float16* __restrict__ k16, float* __restrict__ S) {
  int wave = (blockIdx.x * blockDim.x + threadIdx.x) >> 5;
  int lane = threadIdx.x & 31;
  if (wave >= NB * 12 * 12) return;
  int b = wave / 144;
  int t = wave % 144;
  int tm = t / 12, tn = t % 12;
  const _Float16* qb = q16 + (size_t)b * LT * DINC;
  const _Float16* kb = k16 + (size_t)b * LT * DINC;
  int r = lane & 15, hi = lane >> 4;
  v8f acc = {};
  for (int k0 = 0; k0 < LT; k0 += 32) {
    if (k0 + 64 < LT) {   // stream two steps ahead -> global_prefetch_b8
      __builtin_prefetch(qb + (size_t)(k0 + 64) * DINC, 0, 0);
      __builtin_prefetch(kb + (size_t)(k0 + 64) * DINC, 0, 0);
    }
    v16h a;   // A = q^T : A[m][kl] = q[kl][m]
#pragma unroll
    for (int j = 0; j < 16; ++j)
      a[j] = qb[(size_t)(k0 + kmap(j, hi)) * DINC + tm * 16 + r];
    v16h bf = frag_b_km(kb, DINC, k0, tn * 16, lane);
    acc = wmma16(a, bf, acc);
  }
  int n = lane & 15;
#pragma unroll
  for (int v = 0; v < 8; ++v)
    S[(size_t)b * DINC * DINC + (size_t)(tm * 16 + v + hi * 8) * DINC + tn * 16 + n] =
        acc[v];
}

// ---------------- softmax over last axis (192) , one warp / row -------------
__global__ void k_softmax(const float* __restrict__ S, _Float16* __restrict__ attn16) {
  int warp = (blockIdx.x * blockDim.x + threadIdx.x) >> 5;
  int lane = threadIdx.x & 31;
  if (warp >= NB * DINC) return;
  const float* row = S + (size_t)warp * DINC;
  float vals[6];
  float mx = -1e30f;
#pragma unroll
  for (int i = 0; i < 6; ++i) {
    vals[i] = row[lane + 32 * i];
    mx = fmaxf(mx, vals[i]);
  }
#pragma unroll
  for (int o = 16; o > 0; o >>= 1) mx = fmaxf(mx, __shfl_xor(mx, o));
  float sum = 0.f;
#pragma unroll
  for (int i = 0; i < 6; ++i) {
    vals[i] = __expf(vals[i] - mx);
    sum += vals[i];
  }
#pragma unroll
  for (int o = 16; o > 0; o >>= 1) sum += __shfl_xor(sum, o);
  float inv = 1.f / sum;
  _Float16* orow = attn16 + (size_t)warp * DINC;
#pragma unroll
  for (int i = 0; i < 6; ++i) orow[lane + 32 * i] = (_Float16)(vals[i] * inv);
}

// ---------------- out_a = V (Lx192) @ attn^T (192x192) ----------------------
__global__ void k_gemm_outa(const _Float16* __restrict__ v16f,
                            const _Float16* __restrict__ attn16,
                            _Float16* __restrict__ outa) {
  int wave = (blockIdx.x * blockDim.x + threadIdx.x) >> 5;
  int lane = threadIdx.x & 31;
  const int PT = (LT / 16) * 12;
  if (wave >= NB * PT) return;
  int b = wave / PT;
  int t = wave % PT;
  int tm = t / 12, tn = t % 12;
  const _Float16* vb = v16f + (size_t)b * LT * DINC;
  const _Float16* ab = attn16 + (size_t)b * DINC * DINC;
  v8f acc = {};
#pragma unroll
  for (int k0 = 0; k0 < DINC; k0 += 32) {
    v16h a = frag_a_rm(vb, DINC, (long)tm * 16, k0, lane);
    v16h bf = frag_b_wt(ab, DINC, tn * 16, k0, lane);
    acc = wmma16(a, bf, acc);
  }
  int n = lane & 15, hi = lane >> 4;
#pragma unroll
  for (int v = 0; v < 8; ++v)
    outa[(size_t)b * LT * DINC + (size_t)(tm * 16 + v + hi * 8) * DINC + tn * 16 + n] =
        (_Float16)acc[v];
}

// ---------------- 3x3x3 conv as implicit WMMA GEMM (192ic -> 96oc) ---------
// Block = 8 M-tiles x 1 oc-tile.  Per-tap 16oc x 192ic weight slice (6 KB) is
// double-buffered in LDS via async global->LDS (ASYNCcnt), pipelined one tap
// ahead; B fragments then come from LDS (ds_load), A gathered from global with
// branchless boundary masking (EXEC stays all-ones for WMMA).
// mode 0: write f16 into conv2in channels 0..95 (+bias)
// mode 1: write fp32 to o32 (+bias +x residual)
__global__ void __launch_bounds__(256)
k_conv3d(const _Float16* __restrict__ in16, const _Float16* __restrict__ wp,
         const float* __restrict__ bias, _Float16* __restrict__ o16,
         float* __restrict__ o32, const float* __restrict__ xres, int mode) {
  __shared__ _Float16 shw[2][16 * DINC];          // 2 x 6 KB
  const int MT = LT / 16;                          // 864 M-tiles per batch
  int bx = blockIdx.x;                             // NB * (MT/8) * 6 blocks exactly
  int tn = bx % 6;
  int g  = bx / 6;
  int tmB = g % (MT / 8);
  int b   = g / (MT / 8);
  int wv   = threadIdx.x >> 5;                     // 0..7 -> M-tile within block
  int lane = threadIdx.x & 31;
  int tm = tmB * 8 + wv;
  int r = lane & 15, hi = lane >> 4, n = lane & 15;
  int p = tm * 16 + r;                             // spatial position
  int w0 = p % EDGE, h0 = (p / EDGE) % EDGE, d0 = p / (EDGE * EDGE);
  const _Float16* ib = in16 + (size_t)b * DINC * LT;
  const _Float16* wt = wp + (size_t)tn * 16 * 27 * DINC;   // this block's 16 oc rows

  // 384 x 16B async transfers stage one tap's weights; tid<128 does a 2nd one
  // (wave-uniform predicate -> no partial-EXEC issue).
  auto stage = [&](int tap, int buf) {
    int e = threadIdx.x;
    {
      int row = e / 24, c16 = e % 24;
      unsigned lo = (unsigned)(uintptr_t)(&shw[buf][0]) + (unsigned)e * 16u;
      const _Float16* gs = wt + (size_t)row * 27 * DINC + (size_t)tap * DINC + c16 * 8;
      asm volatile("global_load_async_to_lds_b128 %0, %1, off"
                   :: "v"(lo), "v"(gs) : "memory");
    }
    if (threadIdx.x < 128) {
      int e2 = 256 + e;
      int row = e2 / 24, c16 = e2 % 24;
      unsigned lo = (unsigned)(uintptr_t)(&shw[buf][0]) + (unsigned)e2 * 16u;
      const _Float16* gs = wt + (size_t)row * 27 * DINC + (size_t)tap * DINC + c16 * 8;
      asm volatile("global_load_async_to_lds_b128 %0, %1, off"
                   :: "v"(lo), "v"(gs) : "memory");
    }
  };

  v8f acc = {};
  stage(0, 0);
  asm volatile("s_wait_asynccnt 0" ::: "memory");
  __syncthreads();
  for (int tap = 0; tap < 27; ++tap) {
    int cur = tap & 1;
    if (tap + 1 < 27) stage(tap + 1, cur ^ 1);     // prefetch next tap's weights
    int kd = tap / 9 - 1, kh = (tap / 3) % 3 - 1, kw = tap % 3 - 1;
    int nd = d0 + kd, nh = h0 + kh, nw = w0 + kw;
    bool valid = ((unsigned)nd < EDGE) & ((unsigned)nh < EDGE) & ((unsigned)nw < EDGE);
    int nidx = valid ? ((nd * EDGE + nh) * EDGE + nw) : 0;   // clamped, masked below
    const _Float16* lw = &shw[cur][0];             // 16 oc x 192 ic, row-major
    for (int ic0 = 0; ic0 < DINC; ic0 += 32) {
      v16h a;
#pragma unroll
      for (int j = 0; j < 16; ++j) {
        _Float16 val = ib[(size_t)(ic0 + kmap(j, hi)) * LT + nidx];
        a[j] = valid ? val : (_Float16)0.f;        // branchless -> EXEC stays full
      }
      v16h bf;
#pragma unroll
      for (int j = 0; j < 16; ++j)
        bf[j] = lw[n * DINC + ic0 + hi * 16 + j];  // ds_load_b128 x2
      acc = wmma16(a, bf, acc);
    }
    asm volatile("s_wait_asynccnt 0" ::: "memory");
    __syncthreads();                               // buffer handoff
  }
#pragma unroll
  for (int v = 0; v < 8; ++v) {
    int row = tm * 16 + v + hi * 8;                // spatial position within batch
    int oc = tn * 16 + n;
    float val = acc[v] + bias[oc];
    if (mode == 0) {
      o16[(size_t)b * DINC * LT + (size_t)oc * LT + row] = (_Float16)val;
    } else {
      size_t oi = (size_t)b * DIMC * LT + (size_t)oc * LT + row;
      o32[oi] = val + xres[oi];
    }
  }
}

// ---------------- host side ----------------
extern "C" void kernel_launch(void* const* d_in, const int* in_sizes, int n_in,
                              void* d_out, int out_size, void* d_ws, size_t ws_size,
                              hipStream_t stream) {
  (void)in_sizes; (void)n_in; (void)out_size; (void)ws_size;
  const float* x        = (const float*)d_in[0];
  const float* ln_g     = (const float*)d_in[1];
  const float* ln_b     = (const float*)d_in[2];
  const float* in_proj  = (const float*)d_in[3];
  const float* conv_w   = (const float*)d_in[4];
  const float* conv_b   = (const float*)d_in[5];
  const float* xproj_w  = (const float*)d_in[6];
  const float* dt_w     = (const float*)d_in[7];
  const float* dt_b     = (const float*)d_in[8];
  const float* A_log    = (const float*)d_in[9];
  const float* Dp       = (const float*)d_in[10];
  const float* out_proj = (const float*)d_in[11];
  const float* f1_w     = (const float*)d_in[12];
  const float* f1_b     = (const float*)d_in[13];
  const float* f2_w     = (const float*)d_in[14];
  const float* f2_b     = (const float*)d_in[15];
  float* out = (float*)d_out;

  char* p = (char*)d_ws;
  auto alloc = [&](size_t bytes) -> char* {
    char* r = p;
    p += (bytes + 255) & ~(size_t)255;
    return r;
  };
  _Float16* xnorm16 = (_Float16*)alloc((size_t)BLT * DIMC * 2);
  float*    xin     = (float*)   alloc((size_t)BLT * DINC * 4);
  float*    gate    = (float*)   alloc((size_t)BLT * DINC * 4);
  float*    u_[3]; _Float16* u16_[3]; float* xdbl_[3]; float* dt_[3]; _Float16* y16_[3];
  for (int i = 0; i < 3; ++i) {
    u_[i]    = (float*)   alloc((size_t)BLT * DINC * 4);
    u16_[i]  = (_Float16*)alloc((size_t)BLT * DINC * 2);
    xdbl_[i] = (float*)   alloc((size_t)BLT * XLD * 4);
    dt_[i]   = (float*)   alloc((size_t)BLT * DINC * 4);
    y16_[i]  = (_Float16*)alloc((size_t)BLT * DINC * 2);
  }
  _Float16* yS16    = (_Float16*)alloc((size_t)BLT * DINC * 2);
  float*    S       = (float*)   alloc((size_t)NB * DINC * DINC * 4);
  _Float16* attn16  = (_Float16*)alloc((size_t)NB * DINC * DINC * 2);
  _Float16* outa16  = (_Float16*)alloc((size_t)NB * LT * DINC * 2);
  _Float16* conv2in = (_Float16*)alloc((size_t)NB * DINC * LT * 2);
  _Float16* inW16   = (_Float16*)alloc((size_t)2 * DINC * 2 * DIMC * 2);
  _Float16* Wo16    = (_Float16*)alloc((size_t)DIMC * DINC * 2);
  _Float16* xprW16  = (_Float16*)alloc((size_t)3 * XLD * DINC * 2);
  _Float16* wf1     = (_Float16*)alloc((size_t)DIMC * 27 * DINC * 2);
  _Float16* wf2     = (_Float16*)alloc((size_t)DIMC * 27 * DINC * 2);
  float*    aneg    = (float*)   alloc((size_t)3 * DINC * DSN * 4);

  const int TPB = 256;
  auto gb = [](long n) { return (int)((n + 255) / 256); };   // thread-grid blocks
  auto wb = [](long w) { return (int)((w + 7) / 8); };       // wave-grid blocks (8 waves/blk)

  // weight packing
  k_cast_f16 <<<gb(2 * DINC * 2 * DIMC), TPB, 0, stream>>>(in_proj, inW16, 2 * DINC * 2 * DIMC);
  k_cast_f16 <<<gb(DIMC * DINC), TPB, 0, stream>>>(out_proj, Wo16, DIMC * DINC);
  k_pack_xproj<<<gb(3 * XLD * DINC), TPB, 0, stream>>>(xproj_w, xprW16);
  k_pack_convw<<<gb(DIMC * 27 * DINC), TPB, 0, stream>>>(f1_w, wf1);
  k_pack_convw<<<gb(DIMC * 27 * DINC), TPB, 0, stream>>>(f2_w, wf2);
  k_pack_aneg <<<gb(3 * DINC * DSN), TPB, 0, stream>>>(A_log, aneg);

  // trunk
  k_layernorm<<<gb(BLT), TPB, 0, stream>>>(x, ln_g, ln_b, xnorm16);
  k_gemm_inproj<<<wb((long)(BLT / 16) * 24), TPB, 0, stream>>>(xnorm16, inW16, xin, gate);

  for (int br = 0; br < 3; ++br) {
    k_conv1d<<<gb((long)BLT * DINC), TPB, 0, stream>>>(xin, conv_w, conv_b, u_[br],
                                                       u16_[br], br);
    k_gemm_xproj<<<wb((long)(BLT / 16) * 3), TPB, 0, stream>>>(
        u16_[br], xprW16 + (size_t)br * XLD * DINC, xdbl_[br]);
    k_dtproj<<<gb((long)BLT * DINC), TPB, 0, stream>>>(xdbl_[br], dt_w, dt_b, dt_[br], br);
    k_scan<<<wb((long)NB * DINC / 2), TPB, 0, stream>>>(u_[br], dt_[br], xdbl_[br], aneg,
                                                        Dp, gate, y16_[br], br);
  }

  k_sum_qkv<<<gb((long)BLT * DINC), TPB, 0, stream>>>(y16_[0], y16_[1], y16_[2], yS16);
  k_gemm_out<<<wb((long)(BLT / 16) * 6), TPB, 0, stream>>>(yS16, Wo16, conv2in);

  k_gemm_attn<<<wb((long)NB * 144), TPB, 0, stream>>>(y16_[0], y16_[1], S);
  k_softmax<<<gb((long)NB * DINC * 32), TPB, 0, stream>>>(S, attn16);
  k_gemm_outa<<<wb((long)NB * (LT / 16) * 12), TPB, 0, stream>>>(y16_[2], attn16, outa16);

  const int convBlocks = NB * (LT / 16 / 8) * 6;   // 1296, exact
  // conv f1: out_a (reinterpreted channel-major) -> conv2in channels 0..95
  k_conv3d<<<convBlocks, TPB, 0, stream>>>(outa16, wf1, f1_b, conv2in,
                                           (float*)nullptr, (const float*)nullptr, 0);
  // conv f2: concat(192 ch) -> output + residual
  k_conv3d<<<convBlocks, TPB, 0, stream>>>(conv2in, wf2, f2_b,
                                           (_Float16*)nullptr, out, x, 1);
}
